// MultiHeadedAttention_52570399703091
// MI455X (gfx1250) — matrix operand order repaired and checked by reading, not compiled, not run on hardware
//
#include <hip/hip_runtime.h>
#include <hip/hip_bf16.h>

typedef __attribute__((ext_vector_type(16))) _Float16 v16h;
typedef __attribute__((ext_vector_type(8)))  float    v8f;
typedef __attribute__((ext_vector_type(4)))  float    v4f;

union F16Frag { v16h h; v4f f[2]; };

constexpr int S_LEN = 2048;
constexpr int EMB   = 1024;
constexpr int NHEAD = 16;
constexpr int DK    = 64;
constexpr int BATCH = 4;

static __device__ __forceinline__ v8f wmma_f16(v16h a, v16h b, v8f c) {
  // D = A(16x32 f16) * B(32x16 f16) + C(16x16 f32)
  return __builtin_amdgcn_wmma_f32_16x16x32_f16(false, a, false, b, (short)0, c,
                                                false, false);
}

// ---------------------------------------------------------------------------
// Tiled WMMA GEMM with bias:  C[M,N] = A[M,K] * Bw[K,N] + bias[N]
// OUTMODE 0: plain f32 row-major (M x N)
// OUTMODE 1: f16 head-major      out[((b*H+h)*S + s)*DK + d],  row=b*S+s, col=h*DK+d
// OUTMODE 2: f16 head-transposed out[((b*H+h)*DK + d)*S + s]
// Block: 256 threads = 8 waves (2x4); wave tile 32x32; block tile 64x128; BK=32.
// ---------------------------------------------------------------------------
template<int OUTMODE, typename InT, typename OutT>
__global__ __launch_bounds__(256)
void gemm_bias_wmma(const InT* __restrict__ A, const float* __restrict__ Bw,
                    const float* __restrict__ bias, OutT* __restrict__ C,
                    int M, int N, int K)
{
  constexpr int BM = 64, BN = 128, BK = 32;
  __shared__ alignas(16) _Float16 As[BM * BK];   // [m][k]
  __shared__ alignas(16) _Float16 Bs[BN * BK];   // [n][k]  (transposed stage)

  const int tid  = threadIdx.x;
  const int lane = tid & 31;
  const int wid  = tid >> 5;
  const int wm   = wid >> 2;         // 0..1
  const int wn   = wid & 3;          // 0..3
  const int hlf  = lane >> 4;        // 0..1
  const int l16  = lane & 15;

  const int bm = blockIdx.x * BM;
  const int bn = blockIdx.y * BN;

  v8f acc[2][2];
#pragma unroll
  for (int i = 0; i < 2; ++i)
#pragma unroll
    for (int j = 0; j < 2; ++j)
#pragma unroll
      for (int r = 0; r < 8; ++r) acc[i][j][r] = 0.0f;

  const int arow = tid >> 2;          // 0..63
  const int acol = (tid & 3) * 8;     // 0,8,16,24

  for (int k0 = 0; k0 < K; k0 += BK) {
    // --- stage A tile (64x32), f32/f16 -> f16 ---
#pragma unroll
    for (int i = 0; i < 8; ++i) {
      float v = (float)A[(size_t)(bm + arow) * K + (k0 + acol + i)];
      As[arow * BK + acol + i] = (_Float16)v;
    }
    // --- stage B tile (32x128) transposed into Bs[n][k] ---
#pragma unroll
    for (int g = 0; g < 4; ++g) {
      int e  = tid + g * 256;          // 0..1023
      int kr = e >> 5;                 // 0..31
      int gn = (e & 31) * 4;           // 0..124
#pragma unroll
      for (int j = 0; j < 4; ++j) {
        int col = bn + gn + j;
        float v = (col < N) ? Bw[(size_t)(k0 + kr) * N + col] : 0.0f;
        Bs[(gn + j) * BK + kr] = (_Float16)v;
      }
    }
    __syncthreads();

    // --- fragments + WMMA ---
    v16h af[2], bf[2];
#pragma unroll
    for (int i = 0; i < 2; ++i) {
      int m = wm * 32 + i * 16 + l16;
      F16Frag t;
      int base = hlf * 8;                                 // interleaved A layout
      t.f[0] = *(const v4f*)&As[m * BK + base];           // K = base..base+7
      t.f[1] = *(const v4f*)&As[m * BK + base + 16];      // K = base+16..base+23
      af[i] = t.h;
    }
#pragma unroll
    for (int j = 0; j < 2; ++j) {
      int n = wn * 32 + j * 16 + l16;
      F16Frag t;
      int kb = hlf * 8;                                   // interleaved B layout
      t.f[0] = *(const v4f*)&Bs[n * BK + kb];
      t.f[1] = *(const v4f*)&Bs[n * BK + kb + 16];
      bf[j] = t.h;
    }
#pragma unroll
    for (int i = 0; i < 2; ++i)
#pragma unroll
      for (int j = 0; j < 2; ++j)
        acc[i][j] = wmma_f16(af[i], bf[j], acc[i][j]);

    __syncthreads();
  }

  // --- epilogue: bias + store ---
#pragma unroll
  for (int i = 0; i < 2; ++i) {
#pragma unroll
    for (int j = 0; j < 2; ++j) {
      int col = bn + wn * 32 + j * 16 + l16;
      if (col < N) {
        float bv = bias[col];
#pragma unroll
        for (int r = 0; r < 8; ++r) {
          int gm = bm + wm * 32 + i * 16 + r + 8 * hlf;
          float val = acc[i][j][r] + bv;
          if constexpr (OUTMODE == 0) {
            C[(size_t)gm * N + col] = (OutT)val;
          } else {
            int b = gm >> 11, s = gm & (S_LEN - 1);       // S_LEN = 2048
            int h = col >> 6, d = col & (DK - 1);         // DK = 64
            if constexpr (OUTMODE == 1)
              C[(((size_t)(b * NHEAD + h)) * S_LEN + s) * DK + d] = (OutT)val;
            else
              C[(((size_t)(b * NHEAD + h)) * DK + d) * S_LEN + s] = (OutT)val;
          }
        }
      }
    }
  }
}

// ---------------------------------------------------------------------------
// Flash attention: softmax(Q K^T / sqrt(DK)) V   per (batch, head)
// Q,K: (B,H,S,DK) f16 ; Vt: (B,H,DK,S) f16 ; Ctx out: (B,S,E) f16
// Block = 128 threads (4 waves); each wave owns 16 query rows; key blocks of 32.
// ---------------------------------------------------------------------------
__global__ __launch_bounds__(128)
void attn_fmha(const _Float16* __restrict__ Q, const _Float16* __restrict__ Kh,
               const _Float16* __restrict__ Vt, _Float16* __restrict__ Ctx)
{
  __shared__ alignas(16) _Float16 Ps[4 * 16 * 32];   // per-wave P scratch

  const int lane = threadIdx.x & 31;
  const int wid  = threadIdx.x >> 5;
  const int hlf  = lane >> 4;
  const int l16  = lane & 15;

  const int bh = blockIdx.x;             // 0..63
  const int b  = bh >> 4, h = bh & 15;
  const int q0 = blockIdx.y * 64 + wid * 16;

  const _Float16* Qb = Q  + (size_t)bh * S_LEN * DK;
  const _Float16* Kb = Kh + (size_t)bh * S_LEN * DK;
  const _Float16* Vb = Vt + (size_t)bh * DK * S_LEN;
  _Float16* Pw = &Ps[wid * 16 * 32];

  // Q fragments for this wave's 16 rows (2 chunks of 32 along d)
  v16h aq[2];
  {
    int m = q0 + l16, base = hlf * 8;
#pragma unroll
    for (int c = 0; c < 2; ++c) {
      F16Frag t;
      t.f[0] = *(const v4f*)&Qb[(size_t)m * DK + c * 32 + base];
      t.f[1] = *(const v4f*)&Qb[(size_t)m * DK + c * 32 + base + 16];
      aq[c] = t.h;
    }
  }

  v8f acc[4];
#pragma unroll
  for (int t = 0; t < 4; ++t)
#pragma unroll
    for (int r = 0; r < 8; ++r) acc[t][r] = 0.0f;

  float mrow[8], lrow[8];
#pragma unroll
  for (int r = 0; r < 8; ++r) { mrow[r] = -1e30f; lrow[r] = 0.0f; }
  const float scale = 0.125f;            // 1/sqrt(64)

  for (int kb = 0; kb < S_LEN; kb += 32) {
    // ---- scores: Q(16x64) x K_blk^T(64x32) -> s0 (cols kb..+15), s1 (+16..31)
    v8f s0, s1;
#pragma unroll
    for (int r = 0; r < 8; ++r) { s0[r] = 0.0f; s1[r] = 0.0f; }
#pragma unroll
    for (int c = 0; c < 2; ++c) {
      int kd = c * 32 + hlf * 8;         // interleaved-K B fragment along d
      F16Frag b0, b1;
      b0.f[0] = *(const v4f*)&Kb[(size_t)(kb + l16) * DK + kd];
      b0.f[1] = *(const v4f*)&Kb[(size_t)(kb + l16) * DK + kd + 16];
      b1.f[0] = *(const v4f*)&Kb[(size_t)(kb + 16 + l16) * DK + kd];
      b1.f[1] = *(const v4f*)&Kb[(size_t)(kb + 16 + l16) * DK + kd + 16];
      s0 = wmma_f16(aq[c], b0.h, s0);
      s1 = wmma_f16(aq[c], b1.h, s1);
    }

    // ---- online softmax (row stats across the 16-lane half via xor shuffles)
    float p0[8], p1[8], alpha[8];
#pragma unroll
    for (int r = 0; r < 8; ++r) {
      float a0 = s0[r] * scale, a1 = s1[r] * scale;
      float mx = fmaxf(a0, a1);
      mx = fmaxf(mx, __shfl_xor(mx, 1, 32));
      mx = fmaxf(mx, __shfl_xor(mx, 2, 32));
      mx = fmaxf(mx, __shfl_xor(mx, 4, 32));
      mx = fmaxf(mx, __shfl_xor(mx, 8, 32));
      float mn = fmaxf(mrow[r], mx);
      alpha[r] = __expf(mrow[r] - mn);
      p0[r] = __expf(a0 - mn);
      p1[r] = __expf(a1 - mn);
      float sm = p0[r] + p1[r];
      sm += __shfl_xor(sm, 1, 32);
      sm += __shfl_xor(sm, 2, 32);
      sm += __shfl_xor(sm, 4, 32);
      sm += __shfl_xor(sm, 8, 32);
      lrow[r] = lrow[r] * alpha[r] + sm;
      mrow[r] = mn;
    }
#pragma unroll
    for (int t = 0; t < 4; ++t)
#pragma unroll
      for (int r = 0; r < 8; ++r) acc[t][r] *= alpha[r];

    // ---- re-layout P: C-fragment -> LDS row-major -> A-fragment (same wave)
#pragma unroll
    for (int r = 0; r < 8; ++r) {
      int row = r + 8 * hlf;
      Pw[row * 32 + l16]      = (_Float16)p0[r];
      Pw[row * 32 + 16 + l16] = (_Float16)p1[r];
    }
    F16Frag ap;
    {
      int base = hlf * 8;
      ap.f[0] = *(const v4f*)&Pw[l16 * 32 + base];
      ap.f[1] = *(const v4f*)&Pw[l16 * 32 + base + 16];
    }

    // ---- ctx += P(16x32) x V_blk(32x64); V transposed layout -> direct B frags
#pragma unroll
    for (int t = 0; t < 4; ++t) {
      int d = t * 16 + l16;
      const _Float16* vp = &Vb[(size_t)d * S_LEN + kb + hlf * 8];
      F16Frag bv;
      bv.f[0] = *(const v4f*)vp;
      bv.f[1] = *(const v4f*)(vp + 16);
      acc[t] = wmma_f16(ap.h, bv.h, acc[t]);
    }
  }

  // ---- normalize and write ctx in (B,S,E) f16 layout
#pragma unroll
  for (int t = 0; t < 4; ++t) {
#pragma unroll
    for (int r = 0; r < 8; ++r) {
      int s = q0 + r + 8 * hlf;
      int d = t * 16 + l16;
      float v = acc[t][r] / lrow[r];
      Ctx[((size_t)(b * S_LEN + s)) * EMB + h * DK + d] = (_Float16)v;
    }
  }
}

// ---------------------------------------------------------------------------
extern "C" void kernel_launch(void* const* d_in, const int* in_sizes, int n_in,
                              void* d_out, int out_size, void* d_ws, size_t ws_size,
                              hipStream_t stream) {
  const float* query = (const float*)d_in[0];
  const float* key_  = (const float*)d_in[1];
  const float* value = (const float*)d_in[2];
  const float* Wq = (const float*)d_in[3];
  const float* bq = (const float*)d_in[4];
  const float* Wk = (const float*)d_in[5];
  const float* bk = (const float*)d_in[6];
  const float* Wv = (const float*)d_in[7];
  const float* bv = (const float*)d_in[8];
  const float* Wo = (const float*)d_in[9];
  const float* bo = (const float*)d_in[10];
  float* out = (float*)d_out;

  char* ws = (char*)d_ws;
  const size_t seg = (size_t)BATCH * S_LEN * EMB * sizeof(_Float16);  // 16 MiB
  _Float16* Qf = (_Float16*)(ws);            // (B,H,S,DK)
  _Float16* Kf = (_Float16*)(ws + seg);      // (B,H,S,DK)
  _Float16* Vt = (_Float16*)(ws + 2 * seg);  // (B,H,DK,S)
  _Float16* Cx = (_Float16*)(ws + 3 * seg);  // (B,S,E)

  const int M = BATCH * S_LEN;               // 8192

  dim3 gp(M / 64, EMB / 128);
  gemm_bias_wmma<1, float, _Float16><<<gp, 256, 0, stream>>>(query, Wq, bq, Qf, M, EMB, EMB);
  gemm_bias_wmma<1, float, _Float16><<<gp, 256, 0, stream>>>(key_,  Wk, bk, Kf, M, EMB, EMB);
  gemm_bias_wmma<2, float, _Float16><<<gp, 256, 0, stream>>>(value, Wv, bv, Vt, M, EMB, EMB);

  attn_fmha<<<dim3(BATCH * NHEAD, S_LEN / 64), 128, 0, stream>>>(Qf, Kf, Vt, Cx);

  dim3 go(M / 64, 1);
  gemm_bias_wmma<0, _Float16, float><<<go, 256, 0, stream>>>(Cx, Wo, bo, out, M, DK, EMB);
}
